// MultiHeadAttention_7791070675745
// MI455X (gfx1250) — matrix operand order repaired and checked by reading, not compiled, not run on hardware
//
#include <hip/hip_runtime.h>
#include <cstddef>

// ---------------------------------------------------------------------------
// MHA for MI455X (gfx1250): bf16 WMMA (v_wmma_f32_16x16x32_bf16) everywhere,
// async global->LDS staging (ASYNCcnt) with double buffering.
// Wave32, 256-thread blocks (8 waves).
// ---------------------------------------------------------------------------

typedef __attribute__((ext_vector_type(16))) __bf16 v16bf;
typedef __attribute__((ext_vector_type(8)))  float  v8f;
typedef __attribute__((__vector_size__(16))) int    i32x4;

union Frag16 {          // one 16x32 bf16 WMMA operand per lane: 8 dwords
    v16bf v;
    unsigned int u[8];
};

__device__ __forceinline__ unsigned short f2bf(float f) {
    unsigned int u = __builtin_bit_cast(unsigned int, f);
    unsigned int r = u + 0x7FFFu + ((u >> 16) & 1u);   // round-nearest-even
    return (unsigned short)(r >> 16);
}

// ---------------------------------------------------------------------------
// Async global->LDS b128 copy (GLOBAL_LOAD_ASYNC_TO_LDS_B128, ASYNCcnt) with
// synchronous fallback if the builtin is not available on this toolchain.
// Builtin signature (per clang diagnostic): (i32x4 as1*, i32x4 as3*, Ii, Ii).
// ---------------------------------------------------------------------------
#if __has_builtin(__builtin_amdgcn_global_load_async_to_lds_b128) && \
    __has_builtin(__builtin_amdgcn_s_wait_asynccnt)
#define USE_ASYNC_LDS 1
#endif

#ifdef USE_ASYNC_LDS
__device__ __forceinline__ void async_b128(void* lds, const void* g) {
    __builtin_amdgcn_global_load_async_to_lds_b128(
        (__attribute__((address_space(1))) i32x4*)(unsigned long long)(g),
        (__attribute__((address_space(3))) i32x4*)(unsigned int)(unsigned long long)(lds),
        0, 0);
}
#define ASYNC_WAIT(n) __builtin_amdgcn_s_wait_asynccnt(n)
#else
__device__ __forceinline__ void async_b128(void* lds, const void* g) {
    *reinterpret_cast<uint4*>(lds) = *reinterpret_cast<const uint4*>(g);
}
#define ASYNC_WAIT(n) ((void)0)
#endif

// ---------------------------------------------------------------------------
// fp32 -> bf16 elementwise (vectorized x8)
// ---------------------------------------------------------------------------
__global__ __launch_bounds__(256) void cvt_f32_bf16(const float* __restrict__ in,
                                                    unsigned short* __restrict__ out,
                                                    int n) {
    int i = (blockIdx.x * 256 + threadIdx.x) * 8;
    if (i < n) {
        float4 a = *reinterpret_cast<const float4*>(in + i);
        float4 b = *reinterpret_cast<const float4*>(in + i + 4);
        unsigned short o[8] = {f2bf(a.x), f2bf(a.y), f2bf(a.z), f2bf(a.w),
                               f2bf(b.x), f2bf(b.y), f2bf(b.z), f2bf(b.w)};
        *reinterpret_cast<uint4*>(out + i) = *reinterpret_cast<const uint4*>(o);
    }
}

// fp32 W[K,N] -> bf16 W^T[N,K] (one-time; makes GEMM B staging contiguous)
__global__ __launch_bounds__(256) void cvt_transpose_bf16(const float* __restrict__ in,
                                                          unsigned short* __restrict__ out,
                                                          int K, int N) {
    int i = blockIdx.x * 256 + threadIdx.x;      // over N*K outputs
    if (i < N * K) {
        int n = i / K, k = i - n * K;
        out[i] = f2bf(in[(size_t)k * N + n]);
    }
}

// ---------------------------------------------------------------------------
// GEMM: C[M,N] = A[M,K] * B[K,N], with B supplied PRE-TRANSPOSED as Bt[N,K].
// bf16 in, f32 accumulate. Optional bf16 out (normal or per-head-transposed
// V layout [B,H,64,S]) and optional f32 out with bias.
// Tile: BM=128, BN=128, BK=32. 8 waves 4(M)x2(N); wave tile 32x64 (2x4 WMMA).
// Double-buffered LDS fed by async global->LDS copies.
// ---------------------------------------------------------------------------
#define BM 128
#define BN 128
#define BK 32

__global__ __launch_bounds__(256) void gemm_bf16(
    const unsigned short* __restrict__ A,    // [M,K]
    const unsigned short* __restrict__ Bt,   // [N,K]  (pre-transposed B)
    unsigned short* __restrict__ Cb,         // bf16 out (may be null)
    float* __restrict__ Cf,                  // f32 out (may be null)
    const float* __restrict__ bias,          // may be null
    int M, int N, int K,
    int vtS)                                 // >0: store Cb transposed per head
{
    __shared__ __align__(16) unsigned short As[2][BM * BK];  // [row][k] 2x8KB
    __shared__ __align__(16) unsigned short Bs[2][BN * BK];  // [n][k]   2x8KB

    const int t     = threadIdx.x;
    const int lane  = t & 31;
    const int wave  = t >> 5;
    const int waveM = wave >> 1;     // 0..3
    const int waveN = wave & 1;      // 0..1
    const int half  = lane >> 4;
    const int l16   = lane & 15;

    const int blockM = blockIdx.y * BM;
    const int blockN = blockIdx.x * BN;

    // staging map: thread copies 16 elems of A tile and 16 of Bt tile
    const int srow = t >> 1;             // 0..127
    const int scol = (t & 1) * 16;       // 0 or 16

    const unsigned short* gA = A  + (size_t)(blockM + srow) * K + scol;
    const unsigned short* gB = Bt + (size_t)(blockN + srow) * K + scol;
    unsigned short* lA = &As[0][srow * BK + scol];
    unsigned short* lB = &Bs[0][srow * BK + scol];
    const int lbuf = BM * BK;            // elements per LDS buffer

    v8f acc[2][4] = {};

    const int nsteps = K / BK;
    // prologue: stage k-step 0 into buffer 0  (4 async ops/thread per batch)
    async_b128(lA,     gA);
    async_b128(lA + 8, gA + 8);
    async_b128(lB,     gB);
    async_b128(lB + 8, gB + 8);

    for (int step = 0; step < nsteps; ++step) {
        const int cur = step & 1;
        if (step + 1 < nsteps) {
            const int k1 = (step + 1) * BK;
            const int nb = (cur ^ 1) * lbuf;
            async_b128(lA + nb,     gA + k1);
            async_b128(lA + nb + 8, gA + k1 + 8);
            async_b128(lB + nb,     gB + k1);
            async_b128(lB + nb + 8, gB + k1 + 8);
            ASYNC_WAIT(4);               // wait only for current tile's batch
        } else {
            ASYNC_WAIT(0);
        }
        __syncthreads();

        const unsigned int* AsU = reinterpret_cast<const unsigned int*>(As[cur]);
        const unsigned int* BsU = reinterpret_cast<const unsigned int*>(Bs[cur]);

        // A fragments (ISA 16-bit A layout)
        Frag16 a[2];
#pragma unroll
        for (int mt = 0; mt < 2; ++mt) {
            int row = waveM * 32 + mt * 16 + l16;
#pragma unroll
            for (int r = 0; r < 8; ++r) {
                int kk = ((r < 4) ? 0 : 16) + half * 8 + 2 * (r & 3);
                a[mt].u[r] = AsU[row * (BK / 2) + (kk >> 1)];
            }
        }
#pragma unroll
        for (int nt = 0; nt < 4; ++nt) {
            Frag16 b;
            int n = waveN * 64 + nt * 16 + l16;
#pragma unroll
            for (int j = 0; j < 8; ++j) {
                int kk = ((j < 4) ? 0 : 16) + half * 8 + 2 * (j & 3);
                b.u[j] = BsU[n * (BK / 2) + (kk >> 1)];
            }
#pragma unroll
            for (int mt = 0; mt < 2; ++mt) {
                acc[mt][nt] = __builtin_amdgcn_wmma_f32_16x16x32_bf16(
                    false, a[mt].v, false, b.v, (short)0, acc[mt][nt], false, false);
            }
        }
        __syncthreads();                 // reads done before next overwrite
    }

    // --- epilogue (C/D layout: vgpr r -> M = r + 8*half, lane l16 -> N)
#pragma unroll
    for (int mt = 0; mt < 2; ++mt) {
#pragma unroll
        for (int nt = 0; nt < 4; ++nt) {
            int col = blockN + waveN * 64 + nt * 16 + l16;
            float bv = bias ? bias[col] : 0.0f;
#pragma unroll
            for (int r = 0; r < 8; ++r) {
                int row = blockM + waveM * 32 + mt * 16 + half * 8 + r;
                float val = acc[mt][nt][r] + bv;
                if (Cf) Cf[(size_t)row * N + col] = val;
                if (Cb) {
                    if (vtS > 0) {       // V layout: [B, H, 64, S]
                        int b_ = row / vtS, s_ = row - b_ * vtS;
                        int h_ = col >> 6, d_ = col & 63;
                        Cb[(((size_t)b_ * (N >> 6) + h_) * 64 + d_) * vtS + s_] =
                            f2bf(val);
                    } else {
                        Cb[(size_t)row * N + col] = f2bf(val);
                    }
                }
            }
        }
    }
}

// ---------------------------------------------------------------------------
// Flash attention: block per (b, h, 128 q rows); 8 waves, wave owns 16 rows.
// K staged [key][d], V pre-transposed globally [B,H,64,S] so staging is
// contiguous. Double-buffered async staging; online softmax; bf16 WMMA.
// ---------------------------------------------------------------------------
#define HD 64
#define QT 128
#define KT 32

__global__ __launch_bounds__(256) void attention_fwd(
    const unsigned short* __restrict__ Q,    // [B*S, D] bf16
    const unsigned short* __restrict__ Km,   // [B*S, D] bf16
    const unsigned short* __restrict__ Vt,   // [B, H, 64, S] bf16
    unsigned short* __restrict__ ctx,        // [B*S, D] bf16
    int Bn, int S, int Hn, int Dm)
{
    const int nq   = S / QT;
    const int bid  = blockIdx.x;
    const int qblk = bid % nq;
    const int h    = (bid / nq) % Hn;
    const int b    = bid / (nq * Hn);

    __shared__ __align__(16) unsigned short KL[2][KT * HD];    // [key][d] 2x4KB
    __shared__ __align__(16) unsigned short VL[2][HD * KT];    // [d][key] 2x4KB
    __shared__ __align__(16) unsigned short Ptile[8][16 * KT]; // per-wave  8KB

    const int t    = threadIdx.x;
    const int lane = t & 31;
    const int wave = t >> 5;
    const int half = lane >> 4;
    const int l16  = lane & 15;

    const int    qbase   = qblk * QT;
    const int    qrow0   = qbase + wave * 16;
    const size_t headOff = (size_t)h * HD;

    // staging maps (1 b128 of K + 1 b128 of V per thread per key block)
    const int  kkey = t >> 3, kcol = (t & 7) * 8;          // K: [32][64]
    const int  vrow = t >> 2, vcol = (t & 3) * 8;          // V: [64][32]
    const unsigned short* gK = Km + ((size_t)b * S + kkey) * Dm + headOff + kcol;
    const unsigned short* gV = Vt + (((size_t)b * Hn + h) * HD + vrow) * S + vcol;
    unsigned short* lK = &KL[0][kkey * HD + kcol];
    unsigned short* lV = &VL[0][vrow * KT + vcol];
    const int lKbuf = KT * HD, lVbuf = HD * KT;

    // --- Q fragments: 16x64 -> two 16x32 A operands, straight from global
    Frag16 qf[2];
    {
        const unsigned int* qu = reinterpret_cast<const unsigned int*>(
            Q + ((size_t)b * S + qrow0 + l16) * Dm + headOff);
#pragma unroll
        for (int c = 0; c < 2; ++c)
#pragma unroll
            for (int r = 0; r < 8; ++r) {
                int kk = c * 32 + ((r < 4) ? 0 : 16) + half * 8 + 2 * (r & 3);
                qf[c].u[r] = qu[kk >> 1];
            }
    }

    float mrun[8], lrun[8];
    v8f   o[4] = {};
#pragma unroll
    for (int r = 0; r < 8; ++r) { mrun[r] = -3.0e38f; lrun[r] = 0.0f; }

    const float scale = 0.125f;            // 1/sqrt(64)
    const int   nkblk = (qbase + QT) / KT;

    // prologue: stage key block 0 (2 async ops/thread per batch)
    async_b128(lK, gK);
    async_b128(lV, gV);

    for (int kb = 0; kb < nkblk; ++kb) {
        const int cur   = kb & 1;
        const int kbase = kb * KT;
        if (kb + 1 < nkblk) {
            const size_t adv = (size_t)(kb + 1) * KT;
            async_b128(lK + (cur ^ 1) * lKbuf, gK + adv * Dm);
            async_b128(lV + (cur ^ 1) * lVbuf, gV + adv);
            ASYNC_WAIT(2);
        } else {
            ASYNC_WAIT(0);
        }
        __syncthreads();

        const unsigned int* KU = reinterpret_cast<const unsigned int*>(KL[cur]);
        const unsigned int* VU = reinterpret_cast<const unsigned int*>(VL[cur]);

        // --- scores: 2 key tiles x 2 d-chunks = 4 WMMAs
        v8f sc[2] = {};
#pragma unroll
        for (int nt = 0; nt < 2; ++nt) {
#pragma unroll
            for (int c = 0; c < 2; ++c) {
                Frag16 kf;
                int key = nt * 16 + l16;
#pragma unroll
                for (int j = 0; j < 8; ++j) {
                    int d = c * 32 + ((j < 4) ? 0 : 16) + half * 8 + 2 * (j & 3);
                    kf.u[j] = KU[key * (HD / 2) + (d >> 1)];
                }
                sc[nt] = __builtin_amdgcn_wmma_f32_16x16x32_bf16(
                    false, qf[c].v, false, kf.v, (short)0, sc[nt], false, false);
            }
        }

        // --- scale + causal mask (C layout: row m = r + 8*half, col = l16)
#pragma unroll
        for (int nt = 0; nt < 2; ++nt) {
            int kabs = kbase + nt * 16 + l16;
#pragma unroll
            for (int r = 0; r < 8; ++r) {
                int qabs = qrow0 + half * 8 + r;
                float s = sc[nt][r] * scale;
                sc[nt][r] = (kabs <= qabs) ? s : -3.0e38f;
            }
        }

        // --- online softmax: width-16 shfl row reductions
        float alpha[8];
#pragma unroll
        for (int r = 0; r < 8; ++r) {
            float mx = fmaxf(sc[0][r], sc[1][r]);
#pragma unroll
            for (int off = 8; off >= 1; off >>= 1)
                mx = fmaxf(mx, __shfl_xor(mx, off, 16));
            float mnew = fmaxf(mrun[r], mx);
            alpha[r] = __expf(mrun[r] - mnew);
            mrun[r]  = mnew;
        }
#pragma unroll
        for (int nt = 0; nt < 2; ++nt)
#pragma unroll
            for (int r = 0; r < 8; ++r)
                sc[nt][r] = __expf(sc[nt][r] - mrun[r]);
#pragma unroll
        for (int r = 0; r < 8; ++r) {
            float rs = sc[0][r] + sc[1][r];
#pragma unroll
            for (int off = 8; off >= 1; off >>= 1)
                rs += __shfl_xor(rs, off, 16);
            lrun[r] = lrun[r] * alpha[r] + rs;
        }

        // --- rescale running O
#pragma unroll
        for (int tt = 0; tt < 4; ++tt)
#pragma unroll
            for (int r = 0; r < 8; ++r)
                o[tt][r] *= alpha[r];

        // --- P: C-layout regs -> per-wave LDS tile [16][32] -> A-layout frag
        unsigned short* P = &Ptile[wave][0];
#pragma unroll
        for (int nt = 0; nt < 2; ++nt)
#pragma unroll
            for (int r = 0; r < 8; ++r)
                P[(half * 8 + r) * KT + nt * 16 + l16] = f2bf(sc[nt][r]);

        const unsigned int* PU = reinterpret_cast<const unsigned int*>(P);
        Frag16 pf;
#pragma unroll
        for (int r = 0; r < 8; ++r) {
            int kk = ((r < 4) ? 0 : 16) + half * 8 + 2 * (r & 3);
            pf.u[r] = PU[l16 * (KT / 2) + (kk >> 1)];
        }

        // --- O[tt] += P(16x32) x V(32x16): 4 WMMAs
#pragma unroll
        for (int tt = 0; tt < 4; ++tt) {
            Frag16 vf;
            int dcol = tt * 16 + l16;
#pragma unroll
            for (int j = 0; j < 8; ++j) {
                int kc = ((j < 4) ? 0 : 16) + half * 8 + 2 * (j & 3);
                vf.u[j] = VU[dcol * (KT / 2) + (kc >> 1)];
            }
            o[tt] = __builtin_amdgcn_wmma_f32_16x16x32_bf16(
                false, pf.v, false, vf.v, (short)0, o[tt], false, false);
        }
        __syncthreads();
    }

    // --- finalize: 1/l and store ctx (heads merged back into D)
#pragma unroll
    for (int r = 0; r < 8; ++r) lrun[r] = 1.0f / lrun[r];
#pragma unroll
    for (int tt = 0; tt < 4; ++tt) {
        int dcol = tt * 16 + l16;
#pragma unroll
        for (int r = 0; r < 8; ++r) {
            int qabs = qrow0 + half * 8 + r;
            ctx[((size_t)b * S + qabs) * Dm + headOff + dcol] =
                f2bf(o[tt][r] * lrun[r]);
        }
    }
}

// ---------------------------------------------------------------------------
// Host launcher
// ---------------------------------------------------------------------------
extern "C" void kernel_launch(void* const* d_in, const int* in_sizes, int n_in,
                              void* d_out, int out_size, void* d_ws, size_t ws_size,
                              hipStream_t stream) {
    (void)in_sizes; (void)n_in; (void)out_size; (void)ws_size;

    const float* x  = (const float*)d_in[0];
    const float* Wq = (const float*)d_in[1];
    const float* Wk = (const float*)d_in[2];
    const float* Wv = (const float*)d_in[3];
    const float* Wo = (const float*)d_in[4];
    const float* bo = (const float*)d_in[5];
    float* out = (float*)d_out;

    const int Bn = 4, S = 2048, D = 1024, H = 16;
    const int M  = Bn * S;   // 8192

    char* ws = (char*)d_ws;
    size_t off = 0;
    unsigned short* xb   = (unsigned short*)(ws + off); off += (size_t)M * D * 2;
    unsigned short* Wqt  = (unsigned short*)(ws + off); off += (size_t)D * D * 2;
    unsigned short* Wkt  = (unsigned short*)(ws + off); off += (size_t)D * D * 2;
    unsigned short* Wvt  = (unsigned short*)(ws + off); off += (size_t)D * D * 2;
    unsigned short* Wot  = (unsigned short*)(ws + off); off += (size_t)D * D * 2;
    unsigned short* Qb   = (unsigned short*)(ws + off); off += (size_t)M * D * 2;
    unsigned short* Kb   = (unsigned short*)(ws + off); off += (size_t)M * D * 2;
    unsigned short* Vtb  = (unsigned short*)(ws + off); off += (size_t)M * D * 2;
    unsigned short* ctxb = (unsigned short*)(ws + off); off += (size_t)M * D * 2;

    // 1) fp32 -> bf16 (x) and fp32 -> bf16 transposed (weights)
    cvt_f32_bf16<<<(M * D / 8 + 255) / 256, 256, 0, stream>>>(x, xb, M * D);
    cvt_transpose_bf16<<<(D * D + 255) / 256, 256, 0, stream>>>(Wq, Wqt, D, D);
    cvt_transpose_bf16<<<(D * D + 255) / 256, 256, 0, stream>>>(Wk, Wkt, D, D);
    cvt_transpose_bf16<<<(D * D + 255) / 256, 256, 0, stream>>>(Wv, Wvt, D, D);
    cvt_transpose_bf16<<<(D * D + 255) / 256, 256, 0, stream>>>(Wo, Wot, D, D);

    // 2) Q/K/V projections (V written transposed per head: [B,H,64,S])
    dim3 gg(D / BN, M / BM);   // (8, 64)
    gemm_bf16<<<gg, 256, 0, stream>>>(xb, Wqt, Qb,  nullptr, nullptr, M, D, D, 0);
    gemm_bf16<<<gg, 256, 0, stream>>>(xb, Wkt, Kb,  nullptr, nullptr, M, D, D, 0);
    gemm_bf16<<<gg, 256, 0, stream>>>(xb, Wvt, Vtb, nullptr, nullptr, M, D, D, S);

    // 3) causal flash attention
    attention_fwd<<<Bn * H * (S / QT), 256, 0, stream>>>(Qb, Kb, Vtb, ctxb, Bn, S, H, D);

    // 4) output projection + bias (fp32 out)
    gemm_bf16<<<gg, 256, 0, stream>>>(ctxb, Wot, nullptr, out, bo, M, D, D, 0);
}
